// RAFTStereo_27831388078858
// MI455X (gfx1250) — matrix operand order repaired and checked by reading, not compiled, not run on hardware
//
#include <hip/hip_runtime.h>
#include <math.h>

// ---------------------------------------------------------------------------
// RAFT-Stereo forward for MI455X (gfx1250, wave32, WMMA + TDM).
// GEMM work (correlation volume + every conv) -> v_wmma_f32_16x16x32_bf16,
// double-buffered LDS pipeline; conv weight tiles staged by the Tensor Data
// Mover (tensor_load_to_lds, drained with s_wait_tensorcnt).
// ---------------------------------------------------------------------------

typedef __attribute__((ext_vector_type(8)))  float  v8f;
typedef __attribute__((ext_vector_type(16))) __bf16 v16bf;
typedef __attribute__((ext_vector_type(8)))  __bf16 v8bf;
typedef __attribute__((ext_vector_type(4)))  unsigned int tdm_g0_t;
typedef __attribute__((ext_vector_type(8)))  int tdm_g1_t;
typedef __attribute__((ext_vector_type(4)))  int tdm_g2_t;

#define BM 64
#define BN 64
#define BK 32

#if defined(__has_builtin)
#  if __has_builtin(__builtin_amdgcn_tensor_load_to_lds)
#    define HAVE_TDM 1
#  endif
#endif
#ifndef HAVE_TDM
#  define HAVE_TDM 0
#endif

// Diagnostics via stderr (compile-only loop): which TDM path is active?
#if !HAVE_TDM
#  warning "CDNA5 probe: tensor_load_to_lds builtin NOT available -> manual A staging"
#elif __clang_major__ >= 23
#  warning "CDNA5 probe: using 6-arg tensor_load_to_lds (clang >= 23)"
#else
#  warning "CDNA5 probe: using 5-arg tensor_load_to_lds (clang < 23)"
#endif

static inline int ceilDiv(int a, int b) { return (a + b - 1) / b; }

#if HAVE_TDM
// Issue a 2D TDM tile load: tile (tile0 x tile1) of 2-byte elements from
// gaddr (row stride = stride0 elems) into LDS at lds_byte, row-major.
// D# packing per CDNA5 ISA.
__device__ __forceinline__ void tdm_load_2d(unsigned lds_byte, const void* gaddr,
                                            int dim0, int dim1, int tile0, int tile1,
                                            long long stride0) {
  const unsigned long long ga = (unsigned long long)(uintptr_t)gaddr;
  tdm_g0_t g0;
  g0[0] = 1u;                                   // count=1, user descriptor
  g0[1] = lds_byte;                             // lds_addr [63:32]
  g0[2] = (unsigned)(ga & 0xffffffffu);         // global_addr [95:64]
  g0[3] = (unsigned)((ga >> 32) & 0x01ffffffu)  // global_addr [120:96]
          | (2u << 30);                         // type=2 ("image") [127:126]
  tdm_g1_t g1;
  g1[0] = 0x10000;                                        // data_size=1 (2B)
  g1[1] = (int)((dim0 & 0xffff) << 16);                   // tensor_dim0 lo [63:48]
  g1[2] = (int)(((dim0 >> 16) & 0xffff)                   // tensor_dim0 hi [79:64]
          | ((dim1 & 0xffff) << 16));                     // tensor_dim1 lo [95:80]
  g1[3] = (int)(((dim1 >> 16) & 0xffff)                   // tensor_dim1 hi [111:96]
          | ((tile0 & 0xffff) << 16));                    // tile_dim0 [127:112]
  g1[4] = (int)(tile1 & 0xffff);                          // tile_dim1 [143:128]
  g1[5] = (int)(stride0 & 0xffffffffll);                  // dim0_stride lo
  g1[6] = (int)((stride0 >> 32) & 0xffffll);              // dim0_stride hi
  g1[7] = 0;
  const tdm_g2_t z4 = {0, 0, 0, 0};
#if __clang_major__ >= 23
  const tdm_g1_t z8 = {0, 0, 0, 0, 0, 0, 0, 0};
  __builtin_amdgcn_tensor_load_to_lds(g0, g1, z4, z4, z8, 0);
#else
  __builtin_amdgcn_tensor_load_to_lds(g0, g1, z4, z4, 0);
#endif
}
#endif

// ---------------------------------------------------------------------------
// Implicit-GEMM conv (ksz in {1,3,7}, pad=ksz/2). in: [Cin,H,W] bf16,
// wt: [Cout, ksz*ksz, Cin] bf16 (tap-major), bias f32 (nullable).
// 256 threads = 8 waves (4 M-strips x 2 N-strips); block tile 64x64; each
// wave computes a 16x32 strip = 2 WMMA per chunk. Double-buffered LDS:
// chunk q+1 (TDM weights + manual transposed activations) is staged while
// chunk q is multiplied; one barrier per chunk.
// ---------------------------------------------------------------------------
__global__ void conv_wmma(const __bf16* __restrict__ in,
                          const __bf16* __restrict__ wt,
                          const float* __restrict__ bias,
                          float* __restrict__ outf,
                          __bf16* __restrict__ outb,
                          int Cin, int Cout, int H, int W,
                          int ksz, int act, float scale) {
  __shared__ __bf16 lA[2][BM * BK];
  __shared__ __bf16 lB[2][BN * BK];

  const int tid  = threadIdx.x;
  const int lane = tid & 31;
  const int wave = tid >> 5;
  const int wm   = wave >> 1;   // 0..3
  const int wn   = wave & 1;    // 0..1
  const int half = lane >> 4;   // 0/1
  const int l16  = lane & 15;

  const int mBlock = blockIdx.x * BM;
  const int tilesX = W >> 6;
  const int ty = blockIdx.y / tilesX;
  const int x0 = (blockIdx.y % tilesX) << 6;

  const int pad   = ksz >> 1;
  const int taps  = ksz * ksz;
  const int kch   = (Cin + BK - 1) / BK;
  const int total = taps * kch;
  const int HW    = H * W;

  v8f acc0 = {};
  v8f acc1 = {};

  auto stage = [&](int q, int sbuf) {
    const int t = q / kch;
    const int c = q - t * kch;
    const int kbase = c * BK;
    const int dy = t / ksz - pad;
    const int dx = t % ksz - pad;
    const int yy = ty + dy;
    const bool rok = (unsigned)yy < (unsigned)H;
    // ---- A (weights): fragment-ready [m][k] ----
#if HAVE_TDM
    if (wave == 0) {
      const __bf16* ga = wt + ((size_t)mBlock * taps + t) * (size_t)Cin + kbase;
      int dim1 = Cout - mBlock; if (dim1 < 0) dim1 = 0;
      tdm_load_2d((unsigned)(uintptr_t)&lA[sbuf][0], (const void*)ga,
                  Cin - kbase, dim1, BK, BM, (long long)taps * Cin);
    }
#else
    {
      const int m  = tid >> 2;          // 0..63
      const int k0 = (tid & 3) * 8;     // 0..24
      const int gm = mBlock + m;
      const __bf16* wp = wt + ((size_t)gm * taps + t) * (size_t)Cin + kbase + k0;
#pragma unroll
      for (int i = 0; i < 8; ++i) {
        const int gk = kbase + k0 + i;
        __bf16 v = (__bf16)0.0f;
        if (gm < Cout && gk < Cin) v = wp[i];
        lA[sbuf][m * BK + k0 + i] = v;
      }
    }
#endif
    // ---- B (shifted activations): transposed to [col][k] ----
    {
      const int k    = tid >> 3;        // 0..31
      const int col0 = (tid & 7) * 8;   // 0..56
      const int gk   = kbase + k;
      const bool kok = (gk < Cin) && rok;
      const __bf16* ip = in + (size_t)(kok ? gk : 0) * HW + (size_t)(rok ? yy : 0) * W;
      if (kok) __builtin_prefetch((const void*)(ip + W), 0, 1);
#pragma unroll
      for (int j = 0; j < 8; ++j) {
        const int xx = x0 + col0 + j + dx;
        __bf16 v = (__bf16)0.0f;
        if (kok && (unsigned)xx < (unsigned)W) v = ip[xx];
        lB[sbuf][(col0 + j) * BK + k] = v;
      }
    }
  };

  stage(0, 0);
#if HAVE_TDM
  if (wave == 0) __builtin_amdgcn_s_wait_tensorcnt(0);
#endif
  __syncthreads();

  for (int q = 0; q < total; ++q) {
    const int cur = q & 1;
    if (q + 1 < total) stage(q + 1, cur ^ 1);

    // ---- fragments: element s of lane half h holds k = 16h + s (A and B alike) ----
    const __bf16* A  = lA[cur];
    const __bf16* Bm = lB[cur];
    const int arow = wm * 16 + l16;
    const v8bf a0 = *(const v8bf*)&A[arow * BK + 16 * half];
    const v8bf a1 = *(const v8bf*)&A[arow * BK + 16 * half + 8];
    const v16bf afr = __builtin_shufflevector(a0, a1,
        0,1,2,3,4,5,6,7,8,9,10,11,12,13,14,15);
    const int bcol = wn * 32 + l16;
    const v8bf b00 = *(const v8bf*)&Bm[bcol * BK + 16 * half];
    const v8bf b01 = *(const v8bf*)&Bm[bcol * BK + 16 * half + 8];
    const v16bf bfr0 = __builtin_shufflevector(b00, b01,
        0,1,2,3,4,5,6,7,8,9,10,11,12,13,14,15);
    const v8bf b10 = *(const v8bf*)&Bm[(bcol + 16) * BK + 16 * half];
    const v8bf b11 = *(const v8bf*)&Bm[(bcol + 16) * BK + 16 * half + 8];
    const v16bf bfr1 = __builtin_shufflevector(b10, b11,
        0,1,2,3,4,5,6,7,8,9,10,11,12,13,14,15);

    acc0 = __builtin_amdgcn_wmma_f32_16x16x32_bf16(
        false, afr, false, bfr0, (short)0, acc0, false, false);
    acc1 = __builtin_amdgcn_wmma_f32_16x16x32_bf16(
        false, afr, false, bfr1, (short)0, acc1, false, false);
#if HAVE_TDM
    if (wave == 0) __builtin_amdgcn_s_wait_tensorcnt(0);
#endif
    __syncthreads();
  }

  // ---- epilogue: C layout (lane<16 -> M base 0, lane>=16 -> +8) ----
  const int mrow0 = wm * 16 + half * 8;
  const int gn0   = x0 + wn * 32 + l16;
#pragma unroll
  for (int r = 0; r < 8; ++r) {
    const int gm = mBlock + mrow0 + r;
    if (gm < Cout) {
      float b = bias ? bias[gm] : 0.0f;
      float v0 = (acc0[r] + b) * scale;
      float v1 = (acc1[r] + b) * scale;
      if (act == 1) { v0 = v0 > 0.0f ? v0 : 0.0f; v1 = v1 > 0.0f ? v1 : 0.0f; }
      const size_t o = (size_t)gm * HW + (size_t)ty * W + gn0;
      if (outf) { outf[o] = v0; outf[o + 16] = v1; }
      if (outb) { outb[o] = (__bf16)v0; outb[o + 16] = (__bf16)v1; }
    }
  }
}

// ---------------------------------------------------------------------------
// Correlation volume: corr[h, m, n] = sum_k f2[k,h,m] * f3[k,h,n] / 16.
// grid = (W/64, W/64, H); both operands transposed on LDS staging;
// same double-buffered pipeline.
// ---------------------------------------------------------------------------
__global__ void corr_wmma(const __bf16* __restrict__ f2,
                          const __bf16* __restrict__ f3,
                          float* __restrict__ corr,
                          int D, int H, int W) {
  __shared__ __bf16 lA[2][BM * BK];
  __shared__ __bf16 lB[2][BN * BK];

  const int tid  = threadIdx.x;
  const int lane = tid & 31;
  const int wave = tid >> 5;
  const int wm   = wave >> 1;
  const int wn   = wave & 1;
  const int half = lane >> 4;
  const int l16  = lane & 15;

  const int m0 = blockIdx.x * BM;
  const int n0 = blockIdx.y * BN;
  const int h  = blockIdx.z;
  const int HW = H * W;
  const int chunks = D / BK;

  v8f acc0 = {};
  v8f acc1 = {};

  auto stage = [&](int c, int sbuf) {
    const int kbase = c * BK;
    {
      const int k  = tid >> 3;
      const int mq = (tid & 7) * 8;
      const __bf16* p = f2 + (size_t)(kbase + k) * HW + (size_t)h * W + m0 + mq;
      __builtin_prefetch((const void*)(p + (size_t)BK * HW), 0, 1);
#pragma unroll
      for (int i = 0; i < 8; ++i) lA[sbuf][(mq + i) * BK + k] = p[i];
    }
    {
      const int k    = tid >> 3;
      const int col0 = (tid & 7) * 8;
      const __bf16* p = f3 + (size_t)(kbase + k) * HW + (size_t)h * W + n0 + col0;
#pragma unroll
      for (int j = 0; j < 8; ++j) lB[sbuf][(col0 + j) * BK + k] = p[j];
    }
  };

  stage(0, 0);
  __syncthreads();

  for (int c = 0; c < chunks; ++c) {
    const int cur = c & 1;
    if (c + 1 < chunks) stage(c + 1, cur ^ 1);

    const __bf16* A  = lA[cur];
    const __bf16* Bm = lB[cur];
    const int arow = wm * 16 + l16;
    const v8bf a0 = *(const v8bf*)&A[arow * BK + 16 * half];
    const v8bf a1 = *(const v8bf*)&A[arow * BK + 16 * half + 8];
    const v16bf afr = __builtin_shufflevector(a0, a1,
        0,1,2,3,4,5,6,7,8,9,10,11,12,13,14,15);
    const int bcol = wn * 32 + l16;
    const v8bf b00 = *(const v8bf*)&Bm[bcol * BK + 16 * half];
    const v8bf b01 = *(const v8bf*)&Bm[bcol * BK + 16 * half + 8];
    const v16bf bfr0 = __builtin_shufflevector(b00, b01,
        0,1,2,3,4,5,6,7,8,9,10,11,12,13,14,15);
    const v8bf b10 = *(const v8bf*)&Bm[(bcol + 16) * BK + 16 * half];
    const v8bf b11 = *(const v8bf*)&Bm[(bcol + 16) * BK + 16 * half + 8];
    const v16bf bfr1 = __builtin_shufflevector(b10, b11,
        0,1,2,3,4,5,6,7,8,9,10,11,12,13,14,15);

    acc0 = __builtin_amdgcn_wmma_f32_16x16x32_bf16(
        false, afr, false, bfr0, (short)0, acc0, false, false);
    acc1 = __builtin_amdgcn_wmma_f32_16x16x32_bf16(
        false, afr, false, bfr1, (short)0, acc1, false, false);
    __syncthreads();
  }

  const int mrow0 = wm * 16 + half * 8;
  const int gn0   = n0 + wn * 32 + l16;
#pragma unroll
  for (int r = 0; r < 8; ++r) {
    const int gm = m0 + mrow0 + r;
    const size_t o = ((size_t)h * W + gm) * (size_t)W + gn0;
    corr[o]      = acc0[r] * 0.0625f;
    corr[o + 16] = acc1[r] * 0.0625f;
  }
}

// ---------------------------------------------------------------------------
// Elementwise / data-movement kernels
// ---------------------------------------------------------------------------
__global__ void k_wpack(const float* __restrict__ w, __bf16* __restrict__ o,
                        int O, int I, int ks, int n) {
  int g = blockIdx.x * blockDim.x + threadIdx.x;
  if (g >= n) return;
  const int taps = ks * ks;
  const int oo = g / (taps * I);
  const int rem = g - oo * taps * I;
  const int t = rem / I;
  const int i = rem - t * I;
  const int ky = t / ks, kx = t % ks;
  o[g] = (__bf16)w[(((size_t)oo * I + i) * ks + ky) * ks + kx];
}

__global__ void k_cvt_bf16(const float* __restrict__ s, __bf16* __restrict__ d, int n) {
  int g = blockIdx.x * blockDim.x + threadIdx.x;
  if (g < n) d[g] = (__bf16)s[g];
}

__global__ void k_relu_bf16(const float* __restrict__ s, __bf16* __restrict__ d, int n) {
  int g = blockIdx.x * blockDim.x + threadIdx.x;
  if (g < n) { float v = s[g]; d[g] = (__bf16)(v > 0.0f ? v : 0.0f); }
}

__global__ void k_tanh(const float* __restrict__ s, float* __restrict__ d, int n) {
  int g = blockIdx.x * blockDim.x + threadIdx.x;
  if (g < n) d[g] = tanhf(s[g]);
}

__global__ void k_cxinit(float* __restrict__ cx, int W, int n) {
  int g = blockIdx.x * blockDim.x + threadIdx.x;
  if (g < n) cx[g] = (float)(g % W);
}

__global__ void k_pyrdown(const float* __restrict__ in, float* __restrict__ out,
                          int rows, int Wl, int n) {
  int g = blockIdx.x * blockDim.x + threadIdx.x;
  if (g >= n) return;
  const int r = g / Wl, c = g - r * Wl;
  out[g] = 0.5f * (in[(size_t)r * 2 * Wl + 2 * c] + in[(size_t)r * 2 * Wl + 2 * c + 1]);
}

// avg_pool2d(3, stride=2, pad=1, count_include_pad): [C,H,W] f32 -> bf16
__global__ void k_pool2x(const float* __restrict__ in, __bf16* __restrict__ out,
                         int C, int H, int W, int n) {
  int g = blockIdx.x * blockDim.x + threadIdx.x;
  if (g >= n) return;
  const int Ho = H >> 1, Wo = W >> 1;
  const int c = g / (Ho * Wo);
  const int rem = g - c * Ho * Wo;
  const int oy = rem / Wo, ox = rem - oy * Wo;
  float s = 0.0f;
#pragma unroll
  for (int dy = -1; dy <= 1; ++dy)
#pragma unroll
    for (int dx = -1; dx <= 1; ++dx) {
      const int y = 2 * oy + dy, x = 2 * ox + dx;
      if ((unsigned)y < (unsigned)H && (unsigned)x < (unsigned)W)
        s += in[((size_t)c * H + y) * W + x];
    }
  out[g] = (__bf16)(s * (1.0f / 9.0f));
}

// bilinear align_corners=True: [C,Hi,Wi] f32 -> [C,Ho,Wo] bf16
__global__ void k_interp(const float* __restrict__ in, __bf16* __restrict__ out,
                         int C, int Hi, int Wi, int Ho, int Wo, int n) {
  int g = blockIdx.x * blockDim.x + threadIdx.x;
  if (g >= n) return;
  const int c = g / (Ho * Wo);
  const int rem = g - c * Ho * Wo;
  const int oy = rem / Wo, ox = rem - oy * Wo;
  const float ys = (float)oy * ((float)(Hi - 1) / (float)(Ho - 1));
  const float xs = (float)ox * ((float)(Wi - 1) / (float)(Wo - 1));
  int y0 = (int)floorf(ys); int y1 = y0 + 1 < Hi ? y0 + 1 : Hi - 1;
  int x0 = (int)floorf(xs); int x1 = x0 + 1 < Wi ? x0 + 1 : Wi - 1;
  const float wy = ys - (float)y0, wx = xs - (float)x0;
  const float* p = in + (size_t)c * Hi * Wi;
  const float a = p[(size_t)y0 * Wi + x0] * (1.0f - wy) + p[(size_t)y1 * Wi + x0] * wy;
  const float b = p[(size_t)y0 * Wi + x1] * (1.0f - wy) + p[(size_t)y1 * Wi + x1] * wy;
  out[g] = (__bf16)(a * (1.0f - wx) + b * wx);
}

// 1D bilinear lookup over 4 pyramid levels, radius 4 -> [36, H*W] bf16
__global__ void k_corr_lookup(const float* __restrict__ p0, const float* __restrict__ p1,
                              const float* __restrict__ p2, const float* __restrict__ p3,
                              const float* __restrict__ cx, __bf16* __restrict__ out,
                              int H, int W) {
  int g = blockIdx.x * blockDim.x + threadIdx.x;
  const int N = H * W;
  if (g >= N) return;
  const float cv = cx[g];
  const float* pyr[4] = { p0, p1, p2, p3 };
#pragma unroll
  for (int l = 0; l < 4; ++l) {
    const int Wl = W >> l;
    const float* c = pyr[l] + (size_t)g * Wl;  // row (h, w1) of level l
    const float xc = cv / (float)(1 << l);
#pragma unroll
    for (int d = 0; d < 9; ++d) {
      const float x = xc + (float)(d - 4);
      const float xf = floorf(x);
      const float f = x - xf;
      const int x0 = (int)xf, x1 = x0 + 1;
      const int x0c = x0 < 0 ? 0 : (x0 >= Wl ? Wl - 1 : x0);
      const int x1c = x1 < 0 ? 0 : (x1 >= Wl ? Wl - 1 : x1);
      const float v0 = ((unsigned)x0 < (unsigned)Wl) ? c[x0c] : 0.0f;
      const float v1 = ((unsigned)x1 < (unsigned)Wl) ? c[x1c] : 0.0f;
      out[(size_t)(l * 9 + d) * N + g] = (__bf16)(v0 * (1.0f - f) + v1 * f);
    }
  }
}

// flow = cx - x; write bf16 into motion-encoder flow input [2,N] and into
// hx0 channels 254/255 (mot = concat(me_out[126], flow[2])).
__global__ void k_flow_bf(const float* __restrict__ cx, __bf16* __restrict__ flowb,
                          __bf16* __restrict__ hx0, int W, int n) {
  int g = blockIdx.x * blockDim.x + threadIdx.x;
  if (g >= n) return;
  const float fl = cx[g] - (float)(g % W);
  flowb[g] = (__bf16)fl;
  flowb[(size_t)n + g] = (__bf16)0.0f;
  hx0[(size_t)254 * n + g] = (__bf16)fl;
  hx0[(size_t)255 * n + g] = (__bf16)0.0f;
}

// z = sigmoid(zraw+cz) -> zs ; hx[h-part] = bf16(sigmoid(rraw+cr) * h)
__global__ void k_gru_zr(const float* __restrict__ zraw, const float* __restrict__ rraw,
                         const float* __restrict__ cz, const float* __restrict__ cr,
                         const float* __restrict__ h, float* __restrict__ zs,
                         __bf16* __restrict__ hxh, int n) {
  int g = blockIdx.x * blockDim.x + threadIdx.x;
  if (g >= n) return;
  const float z = 1.0f / (1.0f + expf(-(zraw[g] + cz[g])));
  const float r = 1.0f / (1.0f + expf(-(rraw[g] + cr[g])));
  zs[g] = z;
  hxh[g] = (__bf16)(r * h[g]);
}

// h = (1-z)*h + z*tanh(qraw+cq)
__global__ void k_gru_out(const float* __restrict__ qraw, const float* __restrict__ cq,
                          const float* __restrict__ zs, float* __restrict__ h, int n) {
  int g = blockIdx.x * blockDim.x + threadIdx.x;
  if (g >= n) return;
  const float q = tanhf(qraw[g] + cq[g]);
  const float z = zs[g];
  h[g] = (1.0f - z) * h[g] + z * q;
}

__global__ void k_coords_add(float* __restrict__ cx, const float* __restrict__ d, int n) {
  int g = blockIdx.x * blockDim.x + threadIdx.x;
  if (g < n) cx[g] += d[g];  // only x-component (stereo: dy forced to 0)
}

// convex-combination 4x upsampling of (cx - x); mask: [144, H*W] f32
__global__ void k_upsample(const float* __restrict__ mask, const float* __restrict__ cx,
                           float* __restrict__ out, int H, int W) {
  const int WU = W * 4;
  const int n = H * 4 * WU;
  int g = blockIdx.x * blockDim.x + threadIdx.x;
  if (g >= n) return;
  const int Y = g / WU, X = g - Y * WU;
  const int y = Y >> 2, fy = Y & 3, x = X >> 2, fx = X & 3;
  const int N = H * W;
  const int pix = y * W + x;
  float mv[9];
  float mx = -1e30f;
#pragma unroll
  for (int k = 0; k < 9; ++k) {
    mv[k] = mask[(size_t)(k * 16 + fy * 4 + fx) * N + pix];
    mx = fmaxf(mx, mv[k]);
  }
  float s = 0.0f, a = 0.0f;
#pragma unroll
  for (int k = 0; k < 9; ++k) {
    const float e = expf(mv[k] - mx);
    s += e;
    const int yy = y + k / 3 - 1, xx = x + k % 3 - 1;
    float fl = 0.0f;
    if ((unsigned)yy < (unsigned)H && (unsigned)xx < (unsigned)W)
      fl = 4.0f * (cx[yy * W + xx] - (float)xx);
    a += e * fl;
  }
  out[g] = a / s;
}

// ---------------------------------------------------------------------------
// Host orchestration
// ---------------------------------------------------------------------------
extern "C" void kernel_launch(void* const* d_in, const int* in_sizes, int n_in,
                              void* d_out, int out_size, void* d_ws, size_t ws_size,
                              hipStream_t stream) {
  (void)in_sizes; (void)n_in; (void)out_size; (void)ws_size;

  const int D = 256, H0 = 128, W0 = 256;
  const int H1 = 64, W1 = 128, H2 = 32, W2 = 64;
  const int N0 = H0 * W0, N1 = H1 * W1, N2 = H2 * W2;
  const int ITERS = 8;  // reference setup_inputs(): iters = 8
  const int HUWU = (H0 * 4) * (W0 * 4);

  const float* F_net[3] = { (const float*)d_in[2], (const float*)d_in[3], (const float*)d_in[4] };
  const float* F_ctx[3] = { (const float*)d_in[5], (const float*)d_in[6], (const float*)d_in[7] };

  char* wp = (char*)d_ws;
  auto alloc = [&](size_t bytes) -> void* {
    void* p = (void*)wp;
    wp += (bytes + 255) & ~(size_t)255;
    return p;
  };
  auto EL = [&](int n) { return dim3((unsigned)ceilDiv(n, 256)); };

  // ---- pack all weights to bf16 [O, taps, I] -----------------------------
  auto packW = [&](int idx, int O, int I, int ks) -> const __bf16* {
    const int n = O * ks * ks * I;
    __bf16* dst = (__bf16*)alloc((size_t)n * 2);
    k_wpack<<<EL(n), 256, 0, stream>>>((const float*)d_in[idx], dst, O, I, ks, n);
    return dst;
  };
  const __bf16* Wzqr[3]; const float* Bzqr[3];
  for (int i = 0; i < 3; ++i) { Wzqr[i] = packW(8 + 2 * i, 384, 128, 3); Bzqr[i] = (const float*)d_in[9 + 2 * i]; }
  const __bf16* Wmec1 = packW(14, 64, 36, 1);  const float* Bmec1 = (const float*)d_in[15];
  const __bf16* Wmec2 = packW(16, 64, 64, 3);  const float* Bmec2 = (const float*)d_in[17];
  const __bf16* Wmef1 = packW(18, 64, 2, 7);   const float* Bmef1 = (const float*)d_in[19];
  const __bf16* Wmef2 = packW(20, 64, 64, 3);  const float* Bmef2 = (const float*)d_in[21];
  const __bf16* Wme   = packW(22, 126, 128, 3); const float* Bme  = (const float*)d_in[23];
  const __bf16* Wg[3][3]; const float* Bg[3][3];  // [g08,g16,g32][z,r,q]
  const int gcin[3] = { 384, 384, 256 };
  for (int l = 0; l < 3; ++l)
    for (int gidx = 0; gidx < 3; ++gidx) {
      Wg[l][gidx] = packW(24 + 6 * l + 2 * gidx, 128, gcin[l], 3);
      Bg[l][gidx] = (const float*)d_in[25 + 6 * l + 2 * gidx];
    }
  const __bf16* Wfh1 = packW(42, 256, 128, 3); const float* Bfh1 = (const float*)d_in[43];
  const __bf16* Wfh2 = packW(44, 2, 256, 3);   const float* Bfh2 = (const float*)d_in[45];
  const __bf16* Wmk1 = packW(46, 256, 128, 3); const float* Bmk1 = (const float*)d_in[47];
  const __bf16* Wmk2 = packW(48, 144, 256, 1); const float* Bmk2 = (const float*)d_in[49];

  // ---- workspace buffers -------------------------------------------------
  __bf16* f2b = (__bf16*)alloc((size_t)D * N0 * 2);
  __bf16* f3b = (__bf16*)alloc((size_t)D * N0 * 2);
  float* pyr[4];
  for (int l = 0; l < 4; ++l) pyr[l] = (float*)alloc((size_t)N0 * (W0 >> l) * 4);
  float* czrq[3] = { (float*)alloc((size_t)384 * N0 * 4),
                     (float*)alloc((size_t)384 * N1 * 4),
                     (float*)alloc((size_t)384 * N2 * 4) };
  float* net[3] = { (float*)alloc((size_t)128 * N0 * 4),
                    (float*)alloc((size_t)128 * N1 * 4),
                    (float*)alloc((size_t)128 * N2 * 4) };
  __bf16* hx0 = (__bf16*)alloc((size_t)384 * N0 * 2);
  __bf16* hx1 = (__bf16*)alloc((size_t)384 * N1 * 2);
  __bf16* hx2 = (__bf16*)alloc((size_t)256 * N2 * 2);
  float* g1 = (float*)alloc((size_t)128 * N0 * 4);
  float* g2 = (float*)alloc((size_t)128 * N0 * 4);
  float* zs = (float*)alloc((size_t)128 * N0 * 4);
  __bf16* corrlk = (__bf16*)alloc((size_t)36 * N0 * 2);
  __bf16* flowb  = (__bf16*)alloc((size_t)2 * N0 * 2);
  __bf16* c1b    = (__bf16*)alloc((size_t)64 * N0 * 2);
  __bf16* m1b    = (__bf16*)alloc((size_t)64 * N0 * 2);
  __bf16* mcat   = (__bf16*)alloc((size_t)128 * N0 * 2);
  __bf16* n0b    = (__bf16*)alloc((size_t)128 * N0 * 2);
  __bf16* fh1b   = (__bf16*)alloc((size_t)256 * N0 * 2);
  __bf16* mk1b   = (__bf16*)alloc((size_t)256 * N0 * 2);
  float* dflow   = (float*)alloc((size_t)2 * N0 * 4);
  float* maskb   = (float*)alloc((size_t)144 * N0 * 4);
  float* cx      = (float*)alloc((size_t)N0 * 4);
  __bf16* rctxb  = (__bf16*)alloc((size_t)128 * N0 * 2);

  auto conv = [&](const __bf16* in, const __bf16* wt, const float* bias,
                  float* outf, __bf16* outb, int Cin, int Cout, int H, int W,
                  int ksz, int act, float scale) {
    dim3 grid((unsigned)ceilDiv(Cout, BM), (unsigned)(H * (W / BN)));
    conv_wmma<<<grid, 256, 0, stream>>>(in, wt, bias, outf, outb,
                                        Cin, Cout, H, W, ksz, act, scale);
  };

  // ---- one-time setup ----------------------------------------------------
  k_cvt_bf16<<<EL(D * N0), 256, 0, stream>>>((const float*)d_in[0], f2b, D * N0);
  k_cvt_bf16<<<EL(D * N0), 256, 0, stream>>>((const float*)d_in[1], f3b, D * N0);
  {
    dim3 cg((unsigned)(W0 / BM), (unsigned)(W0 / BN), (unsigned)H0);
    corr_wmma<<<cg, 256, 0, stream>>>(f2b, f3b, pyr[0], D, H0, W0);
  }
  for (int l = 1; l < 4; ++l) {
    const int Wl = W0 >> l, n = N0 * Wl;
    k_pyrdown<<<EL(n), 256, 0, stream>>>(pyr[l - 1], pyr[l], N0, Wl, n);
  }
  const int Nl[3] = { N0, N1, N2 };
  const int Hl[3] = { H0, H1, H2 }, Wl3[3] = { W0, W1, W2 };
  for (int l = 0; l < 3; ++l) {
    k_tanh<<<EL(128 * Nl[l]), 256, 0, stream>>>(F_net[l], net[l], 128 * Nl[l]);
    k_relu_bf16<<<EL(128 * Nl[l]), 256, 0, stream>>>(F_ctx[l], rctxb, 128 * Nl[l]);
    conv(rctxb, Wzqr[l], Bzqr[l], czrq[l], nullptr, 128, 384, Hl[l], Wl3[l], 3, 0, 1.0f);
  }
  k_cxinit<<<EL(N0), 256, 0, stream>>>(cx, W0, N0);

  auto gru = [&](int l, __bf16* hx, int Cin, int H, int W, float* h) {
    const int n = 128 * Nl[l];
    const float* cz = czrq[l];
    const float* cr = czrq[l] + (size_t)128 * Nl[l];
    const float* cq = czrq[l] + (size_t)256 * Nl[l];
    conv(hx, Wg[l][0], Bg[l][0], g1, nullptr, Cin, 128, H, W, 3, 0, 1.0f);  // z
    conv(hx, Wg[l][1], Bg[l][1], g2, nullptr, Cin, 128, H, W, 3, 0, 1.0f);  // r
    k_gru_zr<<<EL(n), 256, 0, stream>>>(g1, g2, cz, cr, h, zs, hx, n);       // hx[h]=r*h
    conv(hx, Wg[l][2], Bg[l][2], g1, nullptr, Cin, 128, H, W, 3, 0, 1.0f);  // q
    k_gru_out<<<EL(n), 256, 0, stream>>>(g1, cq, zs, h, n);
  };

  // ---- iterations --------------------------------------------------------
  for (int it = 0; it < ITERS; ++it) {
    k_corr_lookup<<<EL(N0), 256, 0, stream>>>(pyr[0], pyr[1], pyr[2], pyr[3],
                                              cx, corrlk, H0, W0);
    k_flow_bf<<<EL(N0), 256, 0, stream>>>(cx, flowb, hx0, W0, N0);

    // g32: hx2 = [n2, pool2x(n1)]
    k_cvt_bf16<<<EL(128 * N2), 256, 0, stream>>>(net[2], hx2, 128 * N2);
    k_pool2x<<<EL(128 * N2), 256, 0, stream>>>(net[1], hx2 + (size_t)128 * N2,
                                               128, H1, W1, 128 * N2);
    gru(2, hx2, 256, H2, W2, net[2]);

    // g16: hx1 = [n1, pool2x(n0), interp(n2)]
    k_cvt_bf16<<<EL(128 * N1), 256, 0, stream>>>(net[1], hx1, 128 * N1);
    k_pool2x<<<EL(128 * N1), 256, 0, stream>>>(net[0], hx1 + (size_t)128 * N1,
                                               128, H0, W0, 128 * N1);
    k_interp<<<EL(128 * N1), 256, 0, stream>>>(net[2], hx1 + (size_t)256 * N1,
                                               128, H2, W2, H1, W1, 128 * N1);
    gru(1, hx1, 384, H1, W1, net[1]);

    // motion encoder -> hx0 channels [128..253] (+flow at 254/255 already set)
    conv(corrlk, Wmec1, Bmec1, nullptr, c1b, 36, 64, H0, W0, 1, 1, 1.0f);
    conv(c1b, Wmec2, Bmec2, nullptr, mcat, 64, 64, H0, W0, 3, 1, 1.0f);
    conv(flowb, Wmef1, Bmef1, nullptr, m1b, 2, 64, H0, W0, 7, 1, 1.0f);
    conv(m1b, Wmef2, Bmef2, nullptr, mcat + (size_t)64 * N0, 64, 64, H0, W0, 3, 1, 1.0f);
    conv(mcat, Wme, Bme, nullptr, hx0 + (size_t)128 * N0, 128, 126, H0, W0, 3, 1, 1.0f);

    // g08: hx0 = [n0, mot(128), interp(n1)]
    k_cvt_bf16<<<EL(128 * N0), 256, 0, stream>>>(net[0], hx0, 128 * N0);
    k_interp<<<EL(128 * N0), 256, 0, stream>>>(net[1], hx0 + (size_t)256 * N0,
                                               128, H1, W1, H0, W0, 128 * N0);
    gru(0, hx0, 384, H0, W0, net[0]);

    // heads
    k_cvt_bf16<<<EL(128 * N0), 256, 0, stream>>>(net[0], n0b, 128 * N0);
    conv(n0b, Wfh1, Bfh1, nullptr, fh1b, 128, 256, H0, W0, 3, 1, 1.0f);
    conv(fh1b, Wfh2, Bfh2, dflow, nullptr, 256, 2, H0, W0, 3, 0, 1.0f);
    k_coords_add<<<EL(N0), 256, 0, stream>>>(cx, dflow, N0);
    conv(n0b, Wmk1, Bmk1, nullptr, mk1b, 128, 256, H0, W0, 3, 1, 1.0f);
    conv(mk1b, Wmk2, Bmk2, maskb, nullptr, 256, 144, H0, W0, 1, 0, 0.25f);
    k_upsample<<<EL(HUWU), 256, 0, stream>>>(maskb, cx,
                                             (float*)d_out + (size_t)it * HUWU, H0, W0);
  }
}
